// Model_84353157694068
// MI455X (gfx1250) — matrix operand order repaired and checked by reading, not compiled, not run on hardware
//
#include <hip/hip_runtime.h>
#include <cstdint>

typedef _Float16 v16h __attribute__((ext_vector_type(16)));
typedef _Float16 v8h  __attribute__((ext_vector_type(8)));
typedef float    v8f  __attribute__((ext_vector_type(8)));
typedef int      v4i  __attribute__((ext_vector_type(4)));

#define NN 4096
#define DD 512
#define PANEL (2 * 32 * 256)  // halves per ping-pong panel (32 KB)

#if defined(__gfx1250__) && __has_builtin(__builtin_amdgcn_global_load_async_to_lds_b128) && \
    __has_builtin(__builtin_amdgcn_s_wait_asynccnt)
#define USE_ASYNC_LDS 1
#define WAIT_ASYNC(n) __builtin_amdgcn_s_wait_asynccnt(n)
#else
#define USE_ASYNC_LDS 0
#define WAIT_ASYNC(n) ((void)0)
#endif

// ---------------------------------------------------------------------------
// WMMA fragment helpers (gfx1250 wave32, V_WMMA_F32_16X16X32_F16)
// A 16x32 f16: lane m(0..15)=row; lanes<16 hold K [k0,k0+16),
//              lanes>=16 hold K [k0+16,k0+32) -- contiguous per lane.
// B 32x16 f16: lane n=col (row n of zn_b); lanes<16 K [k0,k0+16),
//              lanes>=16 K [k0+16,k0+32) -- contiguous per lane.
// ---------------------------------------------------------------------------
__device__ __forceinline__ v16h loadA(const _Float16* z, int row, int k0, int h) {
  const _Float16* p = z + row * DD + k0 + h * 16;
  v8h lo = *(const v8h*)p;
  v8h hi = *(const v8h*)(p + 8);
  return __builtin_shufflevector(lo, hi, 0,1,2,3,4,5,6,7,8,9,10,11,12,13,14,15);
}
// B panel in LDS: [mat][32 rows][256 halves]
__device__ __forceinline__ v16h loadB_lds(const _Float16* b, int row, int k, int h) {
  return *(const v16h*)(b + row * 256 + k + h * 16);
}
__device__ __forceinline__ v8f wmma(v16h a, v16h b, v8f c) {
  return __builtin_amdgcn_wmma_f32_16x16x32_f16(false, a, false, b, (short)0, c, false, false);
}
__device__ __forceinline__ v8f zero8() { v8f z{}; return z; }
// C frag: lane<16 -> col=lane, rows v; lane>=16 -> col=lane-16, rows v+8
__device__ __forceinline__ void storeFrag(float* base, v8f c, int coloff, int m, int h) {
#pragma unroll
  for (int v = 0; v < 8; ++v) base[(v + 8 * h) * 36 + coloff + m] = c[v];
}

// Stage one K-half of the B panel (2 mats x 32 rows x 256 halves = 32 KB).
// Async path: GLOBAL_LOAD_ASYNC_TO_LDS_B128, tracked by ASYNCcnt (16 insts/thread).
__device__ __forceinline__ void stage_half(const _Float16* __restrict__ zn1,
                                           const _Float16* __restrict__ zn2,
                                           _Float16* buf, int jc, int kbase, int tid) {
#pragma unroll
  for (int t = 0; t < 16; ++t) {
    int u = tid + t * 128;          // 0..2047 units of 8 halves
    int mat = u >> 10;              // 0 -> zn1, 1 -> zn2
    int row = (u >> 5) & 31;
    int ko = (u & 31) * 8;
    const _Float16* src = (mat ? zn2 : zn1) + (int64_t)(jc + row) * DD + kbase + ko;
    _Float16* dst = buf + mat * (32 * 256) + row * 256 + ko;
#if USE_ASYNC_LDS
    __attribute__((address_space(1))) void* g0 =
        (__attribute__((address_space(1))) void*)const_cast<_Float16*>(src);
    __attribute__((address_space(3))) void* l0 =
        (__attribute__((address_space(3))) void*)dst;
    __builtin_amdgcn_global_load_async_to_lds_b128(
        (__attribute__((address_space(1))) v4i*)g0,
        (__attribute__((address_space(3))) v4i*)l0, 0, 0);
#else
    *(v8h*)dst = *(const v8h*)src;
#endif
  }
}

// ---------------------------------------------------------------------------
// Kernel 1: row-normalize z1,z2 (fp32) -> f16 copies in workspace
// ---------------------------------------------------------------------------
__global__ void normalize_rows(const float* __restrict__ z1, const float* __restrict__ z2,
                               _Float16* __restrict__ o1, _Float16* __restrict__ o2) {
  int row = blockIdx.x * 8 + (threadIdx.x >> 5);
  int lane = threadIdx.x & 31;
  const float* src;
  _Float16* dst;
  if (row < NN) { src = z1 + (int64_t)row * DD;        dst = o1 + (int64_t)row * DD; }
  else          { src = z2 + (int64_t)(row - NN) * DD; dst = o2 + (int64_t)(row - NN) * DD; }
  const float4* p = (const float4*)(src + lane * 16);
  float4 v[4];
  float ssq = 0.f;
#pragma unroll
  for (int i = 0; i < 4; ++i) {
    v[i] = p[i];
    ssq += v[i].x * v[i].x + v[i].y * v[i].y + v[i].z * v[i].z + v[i].w * v[i].w;
  }
#pragma unroll
  for (int off = 16; off > 0; off >>= 1) ssq += __shfl_xor(ssq, off, 32);
  float inv = 1.0f / fmaxf(sqrtf(ssq), 1e-12f);
  float vals[16] = {v[0].x, v[0].y, v[0].z, v[0].w, v[1].x, v[1].y, v[1].z, v[1].w,
                    v[2].x, v[2].y, v[2].z, v[2].w, v[3].x, v[3].y, v[3].z, v[3].w};
  v16h out;
#pragma unroll
  for (int i = 0; i < 16; ++i) out[i] = (_Float16)(vals[i] * inv);
  *(v16h*)(dst + lane * 16) = out;
}

// ---------------------------------------------------------------------------
// Kernel 2: fused Gram (WMMA) + exp + masked row reductions
// Block = 128 threads (4 waves). Wave w handles rows [by*64+16w, +16).
// Col-chunk = 512 columns per blockIdx.x -> 16 sub-tiles of 32 cols, each
// split into 2 K-halves of 256 -> 32 pipeline phases. B panels are streamed
// global->LDS with double buffering (async-to-LDS when available), so the
// WMMA phase overlaps the next panel's fill. adj rows prefetched during GEMM.
// Accumulator vector layout (41 per row):
//  0-2  R11 @ {.8,.64,.72}   3-5  R12   6-8  R21   9-11 R22
//  12-21 MA1..MA10   22-31 MB1..MB10
//  32-37 adj row sums {c,m,f,ca,ma,fa}   38-40 diag(S12) @ {.8,.64,.72}
// ---------------------------------------------------------------------------
__global__ __launch_bounds__(128, 1) void fused_gram(
    const _Float16* __restrict__ zn1, const _Float16* __restrict__ zn2,
    const float* __restrict__ adjc,  const float* __restrict__ adjm,
    const float* __restrict__ adjf,  const float* __restrict__ adjca,
    const float* __restrict__ adjma, const float* __restrict__ adjfa,
    float* __restrict__ part) {
  extern __shared__ _Float16 dynB[];      // 2 ping-pong panels x 32 KB = 64 KB dynamic
  __shared__ float Wl[4 * 4 * 16 * 36];   // per-wave frag staging, 36 KB static
  const int tid = threadIdx.x;
  const int wave = tid >> 5, lane = tid & 31;
  const int m = lane & 15, h = lane >> 4;
  const int r0 = blockIdx.y * 64 + wave * 16;
  const int c0 = blockIdx.x * 512;
  float* W = Wl + wave * (4 * 576);
  const int arow = r0 + m;
  const int64_t rowoff = (int64_t)arow * NN;

  float acc[41];
#pragma unroll
  for (int i = 0; i < 41; ++i) acc[i] = 0.f;

  const float k08 = 1.25f, k064 = 1.5625f, k072 = 1.38888888888888889f;

  // prologue: kick off phase 0
  stage_half(zn1, zn2, dynB, c0, 0, tid);

  v8f c11a = zero8(), c11b = zero8(), c12a = zero8(), c12b = zero8();
  v8f c21a = zero8(), c21b = zero8(), c22a = zero8(), c22b = zero8();

#pragma unroll 1
  for (int p = 0; p < 32; ++p) {
    const int js = p >> 1, kh = p & 1;
    const int jc = c0 + js * 32;
    const int kbase = kh * 256;
    _Float16* cur = dynB + (p & 1) * PANEL;

    __syncthreads();  // everyone done reading the other buffer (phase p-1)
    if (p < 31) {
      const int pn = p + 1;
      stage_half(zn1, zn2, dynB + (pn & 1) * PANEL, c0 + (pn >> 1) * 32, (pn & 1) * 256, tid);
      WAIT_ASYNC(16);  // drain the *older* batch: phase p is resident (this wave)
    } else {
      WAIT_ASYNC(0);
    }
    __syncthreads();  // all waves confirmed -> panel p fully resident

    const int cb = jc + h * 16;
    if (kh == 0) {
      c11a = zero8(); c11b = zero8(); c12a = zero8(); c12b = zero8();
      c21a = zero8(); c21b = zero8(); c22a = zero8(); c22b = zero8();
      // prefetch this sub-tile's adjacency segments while WMMA runs
      __builtin_prefetch(adjc  + rowoff + cb, 0, 1);
      __builtin_prefetch(adjm  + rowoff + cb, 0, 1);
      __builtin_prefetch(adjf  + rowoff + cb, 0, 1);
      __builtin_prefetch(adjca + rowoff + cb, 0, 1);
      __builtin_prefetch(adjma + rowoff + cb, 0, 1);
      __builtin_prefetch(adjfa + rowoff + cb, 0, 1);
    }

    const _Float16* B1 = cur;
    const _Float16* B2 = cur + 32 * 256;
#pragma unroll 2
    for (int kk = 0; kk < 256; kk += 32) {
      v16h a1  = loadA(zn1, arow, kbase + kk, h);
      v16h a2  = loadA(zn2, arow, kbase + kk, h);
      v16h b1a = loadB_lds(B1, m, kk, h);
      v16h b1b = loadB_lds(B1, 16 + m, kk, h);
      v16h b2a = loadB_lds(B2, m, kk, h);
      v16h b2b = loadB_lds(B2, 16 + m, kk, h);
      c11a = wmma(a1, b1a, c11a);  c11b = wmma(a1, b1b, c11b);
      c12a = wmma(a1, b2a, c12a);  c12b = wmma(a1, b2b, c12b);
      c21a = wmma(a2, b1a, c21a);  c21b = wmma(a2, b1b, c21b);
      c22a = wmma(a2, b2a, c22a);  c22b = wmma(a2, b2b, c22b);
    }
    if (kh == 0) continue;

    storeFrag(W + 0 * 576, c11a, 0, m, h); storeFrag(W + 0 * 576, c11b, 16, m, h);
    storeFrag(W + 1 * 576, c12a, 0, m, h); storeFrag(W + 1 * 576, c12b, 16, m, h);
    storeFrag(W + 2 * 576, c21a, 0, m, h); storeFrag(W + 2 * 576, c21b, 16, m, h);
    storeFrag(W + 3 * 576, c22a, 0, m, h); storeFrag(W + 3 * 576, c22b, 16, m, h);
    // Wave-private LDS region; same-wave RAW handled by DScnt waits.

#pragma unroll 1
    for (int q = 0; q < 4; ++q) {
      const int cq = cb + q * 4;
      float4 vc  = *(const float4*)(adjc  + rowoff + cq);
      float4 vm  = *(const float4*)(adjm  + rowoff + cq);
      float4 vf  = *(const float4*)(adjf  + rowoff + cq);
      float4 vca = *(const float4*)(adjca + rowoff + cq);
      float4 vma = *(const float4*)(adjma + rowoff + cq);
      float4 vfa = *(const float4*)(adjfa + rowoff + cq);
      const float ac4[4]  = {vc.x, vc.y, vc.z, vc.w};
      const float am4[4]  = {vm.x, vm.y, vm.z, vm.w};
      const float af4[4]  = {vf.x, vf.y, vf.z, vf.w};
      const float aca4[4] = {vca.x, vca.y, vca.z, vca.w};
      const float ama4[4] = {vma.x, vma.y, vma.z, vma.w};
      const float afa4[4] = {vfa.x, vfa.y, vfa.z, vfa.w};
#pragma unroll
      for (int e = 0; e < 4; ++e) {
        const int cl = h * 16 + q * 4 + e;
        float s11 = W[0 * 576 + m * 36 + cl];
        float s12 = W[1 * 576 + m * 36 + cl];
        float s21 = W[2 * 576 + m * 36 + cl];
        float s22 = W[3 * 576 + m * 36 + cl];
        float e11_08 = __expf(s11 * k08), e11_064 = __expf(s11 * k064), e11_072 = __expf(s11 * k072);
        float e12_08 = __expf(s12 * k08), e12_064 = __expf(s12 * k064), e12_072 = __expf(s12 * k072);
        float e21_08 = __expf(s21 * k08), e21_064 = __expf(s21 * k064), e21_072 = __expf(s21 * k072);
        float e22_08 = __expf(s22 * k08), e22_064 = __expf(s22 * k064), e22_072 = __expf(s22 * k072);
        acc[0] += e11_08; acc[1] += e11_064; acc[2]  += e11_072;
        acc[3] += e12_08; acc[4] += e12_064; acc[5]  += e12_072;
        acc[6] += e21_08; acc[7] += e21_064; acc[8]  += e21_072;
        acc[9] += e22_08; acc[10] += e22_064; acc[11] += e22_072;
        const float wc = ac4[e], wm2 = am4[e], wf2 = af4[e];
        const float wca = aca4[e], wma = ama4[e], wfa = afa4[e];
        acc[12] += wc  * e11_08;   acc[13] += wca * e22_08;
        acc[14] += wm2 * e11_08;   acc[15] += wma * e22_08;
        acc[16] += wf2 * e11_08;   acc[17] += wfa * e22_08;
        acc[18] += wc  * e11_064;  acc[19] += wfa * e22_064;
        acc[20] += wm2 * e11_072;  acc[21] += wfa * e22_072;
        acc[22] += wc  * e12_08;   acc[23] += wca * e21_08;
        acc[24] += wm2 * e12_08;   acc[25] += wma * e21_08;
        acc[26] += wf2 * e12_08;   acc[27] += wfa * e21_08;
        acc[28] += wc  * e12_064;  acc[29] += wfa * e21_064;
        acc[30] += wm2 * e12_072;  acc[31] += wfa * e21_072;
        acc[32] += wc;  acc[33] += wm2; acc[34] += wf2;
        acc[35] += wca; acc[36] += wma; acc[37] += wfa;
        if (arow == cq + e) { acc[38] += e12_08; acc[39] += e12_064; acc[40] += e12_072; }
      }
    }
  }
  // fold col-half h=1 into h=0, then non-atomic per-(chunk,row) partial store
#pragma unroll
  for (int i = 0; i < 41; ++i) acc[i] += __shfl_down(acc[i], 16, 32);
  if (h == 0) {
#pragma unroll
    for (int i = 0; i < 41; ++i)
      part[((int64_t)blockIdx.x * 41 + i) * NN + arow] = acc[i];
  }
}

// ---------------------------------------------------------------------------
// Kernel 3a: fold 8 col-chunk partials, per-row losses, block partial sums
// grid 16 x 256 threads, one row per thread; fully coalesced fold.
// ---------------------------------------------------------------------------
__global__ void row_losses(const float* __restrict__ part, float* __restrict__ bsum) {
  const int tid = threadIdx.x;
  const int i = blockIdx.x * 256 + tid;  // row
  const float e08 = __expf(1.25f), e064 = __expf(1.5625f), e072 = __expf(1.38888888888888889f);
  const int Ra[10] = {0, 9, 0, 9, 0, 9, 1, 10, 2, 11};
  const int Rb[10] = {3, 6, 3, 6, 3, 6, 4, 7, 5, 8};
  const int Aj[10] = {32, 35, 33, 36, 34, 37, 32, 37, 33, 37};
  const int Dg[10] = {38, 38, 38, 38, 38, 38, 39, 39, 40, 40};
  const float Ed[10] = {e08, e08, e08, e08, e08, e08, e064, e064, e072, e072};

  float v[41];
#pragma unroll
  for (int k = 0; k < 41; ++k) {
    float s = 0.f;
#pragma unroll
    for (int j = 0; j < 8; ++j) s += part[((int64_t)j * 41 + k) * NN + i];
    v[k] = s;
  }
  float vals[13];
#pragma unroll
  for (int k = 0; k < 10; ++k) {
    float pos = v[Dg[k]] + v[12 + k] + v[22 + k];
    float den = v[Ra[k]] + v[Rb[k]] - Ed[k];
    float cnt = 2.f * v[Aj[k]] + 1.f;
    vals[k] = __logf(pos / den) / cnt;
  }
  vals[10] = v[32]; vals[11] = v[33]; vals[12] = v[34];

  __shared__ float red[256];
  for (int k = 0; k < 13; ++k) {
    red[tid] = vals[k];
    __syncthreads();
    for (int s = 128; s > 0; s >>= 1) {
      if (tid < s) red[tid] += red[tid + s];
      __syncthreads();
    }
    if (tid == 0) bsum[blockIdx.x * 13 + k] = red[0];
    __syncthreads();
  }
}

// ---------------------------------------------------------------------------
// Kernel 3b: combine 16 block partials -> scalar loss
// ---------------------------------------------------------------------------
__global__ void combine(const float* __restrict__ bsum, float* __restrict__ out) {
  const int tid = threadIdx.x;
  __shared__ float tot[13];
  if (tid < 13) {
    float s = 0.f;
#pragma unroll
    for (int b = 0; b < 16; ++b) s += bsum[b * 13 + tid];
    tot[tid] = s;
  }
  __syncthreads();
  if (tid == 0) {
    float cl[10];
    for (int k = 0; k < 10; ++k) cl[k] = -tot[k] / (float)NN;
    const float A = 0.55f, B = 0.4f;
    float lc  = A * cl[0] + B * cl[1];
    float lm  = A * cl[2] + B * cl[3];
    float lf  = A * cl[4] + B * cl[5];
    float lcf = A * cl[6] + B * cl[7];
    float lmf = A * cl[8] + B * cl[9];
    float sc = tot[10], sm = tot[11], sf = tot[12];
    float totinfo = sc + sm + sf + 1e-08f;
    out[0] = (sc * lc + sm * lm + sf * lf) / totinfo + 0.2f * (lcf + lmf);
  }
}

// ---------------------------------------------------------------------------
extern "C" void kernel_launch(void* const* d_in, const int* in_sizes, int n_in,
                              void* d_out, int out_size, void* d_ws, size_t ws_size,
                              hipStream_t stream) {
  const float* z1    = (const float*)d_in[0];
  const float* z2    = (const float*)d_in[1];
  const float* adjc  = (const float*)d_in[2];
  const float* adjm  = (const float*)d_in[3];
  const float* adjf  = (const float*)d_in[4];
  const float* adjca = (const float*)d_in[5];
  const float* adjma = (const float*)d_in[6];
  const float* adjfa = (const float*)d_in[7];

  _Float16* zn1 = (_Float16*)d_ws;
  _Float16* zn2 = zn1 + (size_t)NN * DD;
  float* part = (float*)((char*)d_ws + 2ull * NN * DD * sizeof(_Float16));  // +8 MiB
  float* bsum = part + 8ull * 41 * NN;                                      // +5.125 MiB
  float* out = (float*)d_out;

  normalize_rows<<<dim3(1024), 256, 0, stream>>>(z1, z2, zn1, zn2);
  fused_gram<<<dim3(8, 64), 128, 2 * PANEL * sizeof(_Float16), stream>>>(
      zn1, zn2, adjc, adjm, adjf, adjca, adjma, adjfa, part);
  row_losses<<<dim3(16), 256, 0, stream>>>(part, bsum);
  combine<<<dim3(1), 64, 0, stream>>>(bsum, out);
}
